// Painn_78847009620147
// MI455X (gfx1250) — matrix operand order repaired and checked by reading, not compiled, not run on hardware
//
#include <hip/hip_runtime.h>
#include <math.h>

#define N_ATOMS 10000
#define NEDGE   256000
#define FDIM    128
#define NRBF_N  20
#define NMOL_N  100
#define NLAYER  3
#define CUT_F   5.0f
#define EPS_F   1e-8f
#define PI_F    3.14159265358979323846f

typedef __attribute__((ext_vector_type(16))) __bf16 v16bf;
typedef __attribute__((ext_vector_type(2)))  __bf16 v2bf;
typedef __attribute__((ext_vector_type(8)))  float  v8f;

struct BF16Frag { uint4 lo, hi; };     // 32 bytes == one v16bf fragment
struct AFrag    { unsigned int u[8]; } ;

static __device__ __forceinline__ float silu_f(float x) {
    return x / (1.0f + __expf(-x));
}

// two fp32 -> one dword holding two bf16 (lo=f0, hi=f1); lowers to v_cvt_pk_bf16_f32
static __device__ __forceinline__ unsigned int pack_bf16(float f0, float f1) {
    v2bf hv;
    hv[0] = (__bf16)f0;
    hv[1] = (__bf16)f1;
    return __builtin_bit_cast(unsigned int, hv);
}

// ---------------------------------------------------------------------------
// Weight pre-swizzle: fp32 [K x Nc] row-major -> bf16 in WMMA B-fragment order.
// Block layout: 512 halves per (ntile, ktile): half index = lane*16 + i with
// lane = c + ((kr>>3)&1)*16, i = (kr&7) + (kr>>4)*8  (ISA 7.12.2 16-bit B layout).
// ---------------------------------------------------------------------------
__global__ void swizzle_w_kernel(const float* __restrict__ W,
                                 unsigned short* __restrict__ out,
                                 int K, int Nc)
{
    int idx = blockIdx.x * blockDim.x + threadIdx.x;
    if (idx >= K * Nc) return;
    int krg = idx / Nc, cg = idx % Nc;
    int nt = cg >> 4, c = cg & 15;
    int kt = krg >> 5, kr = krg & 31;
    int lt = c + (((kr >> 3) & 1) << 4);
    int ih = (kr & 7) + ((kr >> 4) << 3);
    __bf16 h = (__bf16)W[idx];
    out[((size_t)(nt * (K >> 5) + kt) << 9) + lt * 16 + ih] =
        __builtin_bit_cast(unsigned short, h);
}

// ---------------------------------------------------------------------------
// C[M x Nc] = act(A[M x K] @ W + bias); A fp32 row-major, W pre-swizzled bf16.
// Block: 128 threads = 4 waves; wave w owns n-tile blockIdx.y*4+w. No LDS,
// no barriers: A fragment = 4x global b128 + 8 cvt_pk; B fragment = 2x b128.
// Requires M%16==0, Nc%64==0, K in {128,256}.
// ---------------------------------------------------------------------------
template <int K>
__global__ __launch_bounds__(128) void gemm_wmma_bf16(
    const float* __restrict__ A, const unsigned short* __restrict__ Wsw,
    const float* __restrict__ bias, float* __restrict__ C, int Nc, int act)
{
    const int lane  = threadIdx.x & 31;
    const int wv    = threadIdx.x >> 5;
    const int nl    = lane & 15;
    const int kh    = lane >> 4;
    const int m0    = blockIdx.x * 16;
    const int ntile = blockIdx.y * 4 + wv;

    const float* ap = A + (size_t)(m0 + nl) * K + kh * 8;
    const uint4* bp = (const uint4*)(Wsw + ((size_t)ntile * (K >> 5) << 9)) + lane * 2;

    v8f acc = {};
    #pragma unroll
    for (int kt = 0; kt < K / 32; ++kt) {
        float4 a0 = *(const float4*)(ap + kt * 32);
        float4 a1 = *(const float4*)(ap + kt * 32 + 4);
        float4 a2 = *(const float4*)(ap + kt * 32 + 16);
        float4 a3 = *(const float4*)(ap + kt * 32 + 20);
        BF16Frag fb;
        fb.lo = bp[kt * 64];
        fb.hi = bp[kt * 64 + 1];
        AFrag fa;
        fa.u[0] = pack_bf16(a0.x, a0.y); fa.u[1] = pack_bf16(a0.z, a0.w);
        fa.u[2] = pack_bf16(a1.x, a1.y); fa.u[3] = pack_bf16(a1.z, a1.w);
        fa.u[4] = pack_bf16(a2.x, a2.y); fa.u[5] = pack_bf16(a2.z, a2.w);
        fa.u[6] = pack_bf16(a3.x, a3.y); fa.u[7] = pack_bf16(a3.z, a3.w);
        v16bf a = __builtin_bit_cast(v16bf, fa);
        v16bf b = __builtin_bit_cast(v16bf, fb);
        acc = __builtin_amdgcn_wmma_f32_16x16x32_bf16(
            false, a, false, b, (short)0, acc, false, false);
    }

    const int ncol = ntile * 16 + nl;
    const float bv = bias ? bias[ncol] : 0.0f;
    #pragma unroll
    for (int r = 0; r < 8; ++r) {
        int row = m0 + kh * 8 + r;            // D VGPR r: M=r (+8 for hi half)
        float x = acc[r] + bv;
        if (act) x = silu_f(x);
        C[(size_t)row * Nc + ncol] = x;
    }
}

// ---------------------------------------------------------------------------
// CSR construction (once per launch) + per-edge geometry (layer invariant)
// ---------------------------------------------------------------------------
__global__ void zero_int_kernel(int* __restrict__ a, int n) {
    int i = blockIdx.x * blockDim.x + threadIdx.x;
    if (i < n) a[i] = 0;
}

__global__ void hist_kernel(const int* __restrict__ nbrs, int* __restrict__ deg) {
    int e = blockIdx.x * blockDim.x + threadIdx.x;
    if (e < NEDGE) atomicAdd(&deg[nbrs[2 * e]], 1);
}

// single-block exclusive scan over deg[N] -> rowptr[N+1] and cursor[N]
__global__ void scan_kernel(const int* __restrict__ deg,
                            int* __restrict__ rowptr, int* __restrict__ cursor)
{
    __shared__ int tmp[256];
    __shared__ int carry_s;
    const int t = threadIdx.x;
    if (t == 0) carry_s = 0;
    __syncthreads();
    for (int base = 0; base < N_ATOMS; base += 256) {
        int x = (base + t < N_ATOMS) ? deg[base + t] : 0;
        tmp[t] = x;
        __syncthreads();
        for (int off = 1; off < 256; off <<= 1) {
            int y = (t >= off) ? tmp[t - off] : 0;
            __syncthreads();
            tmp[t] += y;
            __syncthreads();
        }
        int incl = tmp[t];
        int cb = carry_s;
        int excl = cb + incl - x;
        if (base + t < N_ATOMS) { rowptr[base + t] = excl; cursor[base + t] = excl; }
        __syncthreads();
        if (t == 255) carry_s = cb + incl;
        __syncthreads();
    }
    if (t == 0) rowptr[N_ATOMS] = carry_s;
}

__global__ void fill_kernel(const int* __restrict__ nbrs,
                            int* __restrict__ cursor, int* __restrict__ eidx)
{
    int e = blockIdx.x * blockDim.x + threadIdx.x;
    if (e >= NEDGE) return;
    int i = nbrs[2 * e];
    int pos = atomicAdd(&cursor[i], 1);
    eidx[pos] = e;
}

__global__ void geom_kernel(const float* __restrict__ xyz, const int* __restrict__ nbrs,
                            float* __restrict__ rbf, float* __restrict__ unitv,
                            float* __restrict__ fcut)
{
    int e = blockIdx.x * blockDim.x + threadIdx.x;
    if (e >= NEDGE) return;
    int i = nbrs[2 * e], j = nbrs[2 * e + 1];
    float dx = xyz[3*j+0] - xyz[3*i+0];
    float dy = xyz[3*j+1] - xyz[3*i+1];
    float dz = xyz[3*j+2] - xyz[3*i+2];
    float d = sqrtf(dx*dx + dy*dy + dz*dz + EPS_F);
    #pragma unroll
    for (int r = 0; r < NRBF_N; ++r)
        rbf[(size_t)e * NRBF_N + r] = sinf((float)(r + 1) * PI_F * d / CUT_F) / d;
    fcut[e] = (d < CUT_F) ? 0.5f * (cosf(PI_F * d / CUT_F) + 1.0f) : 0.0f;
    unitv[3*e+0] = dx / d; unitv[3*e+1] = dy / d; unitv[3*e+2] = dz / d;
}

// ---------------------------------------------------------------------------
// Message gather: one 128-thread block per node i; thread t owns feature t.
// dist_W columns held in 60 registers (constant over the node's edge list);
// accumulation in registers; plain stores, NO atomics.
// ---------------------------------------------------------------------------
__global__ __launch_bounds__(128) void node_gather_kernel(
    const int* __restrict__ rowptr, const int* __restrict__ eidx,
    const int* __restrict__ nbrs,
    const float* __restrict__ rbf, const float* __restrict__ unitv,
    const float* __restrict__ fcut,
    const float* __restrict__ distW,   // [20 x 384] this layer
    const float* __restrict__ distb,   // [384]
    const float* __restrict__ phi,     // [N x 384]
    const float* __restrict__ v,       // [N x 3 x 128]
    float* __restrict__ s,             // [N x 128]  (incremented)
    float* __restrict__ vdel)          // [N x 3 x 128] (overwritten)
{
    const int i = blockIdx.x;
    const int t = threadIdx.x;

    float dw0[NRBF_N], dw1[NRBF_N], dw2[NRBF_N];
    #pragma unroll
    for (int r = 0; r < NRBF_N; ++r) {
        dw0[r] = distW[r * 384 + t];
        dw1[r] = distW[r * 384 + 128 + t];
        dw2[r] = distW[r * 384 + 256 + t];
    }
    const float db0 = distb[t], db1 = distb[128 + t], db2 = distb[256 + t];

    float acc_s = 0.0f, av0 = 0.0f, av1 = 0.0f, av2 = 0.0f;
    const int p0 = rowptr[i], p1 = rowptr[i + 1];
    for (int p = p0; p < p1; ++p) {
        const int e = eidx[p];
        const int j = nbrs[2 * e + 1];
        const float4* rb4 = (const float4*)(rbf + (size_t)e * NRBF_N);
        float4 r0 = rb4[0], r1 = rb4[1], r2 = rb4[2], r3 = rb4[3], r4 = rb4[4];
        float rbl[NRBF_N] = { r0.x,r0.y,r0.z,r0.w, r1.x,r1.y,r1.z,r1.w,
                              r2.x,r2.y,r2.z,r2.w, r3.x,r3.y,r3.z,r3.w,
                              r4.x,r4.y,r4.z,r4.w };
        float w0 = db0, w1 = db1, w2 = db2;
        #pragma unroll
        for (int r = 0; r < NRBF_N; ++r) {
            w0 += rbl[r] * dw0[r];
            w1 += rbl[r] * dw1[r];
            w2 += rbl[r] * dw2[r];
        }
        const float fc = fcut[e];
        w0 *= fc; w1 *= fc; w2 *= fc;
        const float ux = unitv[3*e+0], uy = unitv[3*e+1], uz = unitv[3*e+2];
        const size_t pj = (size_t)j * 384;
        acc_s += phi[pj + t] * w0;
        const float i1 = phi[pj + 128 + t] * w1;
        const float i2 = phi[pj + 256 + t] * w2;
        av0 += i1 * v[pj +       t] + i2 * ux;
        av1 += i1 * v[pj + 128 + t] + i2 * uy;
        av2 += i1 * v[pj + 256 + t] + i2 * uz;
    }
    s[(size_t)i * 128 + t] += acc_s;
    const size_t vi = (size_t)i * 384;
    vdel[vi +       t] = av0;
    vdel[vi + 128 + t] = av1;
    vdel[vi + 256 + t] = av2;
}

// ---------------------------------------------------------------------------
// Elementwise helpers
// ---------------------------------------------------------------------------
__global__ void zero_energy_kernel(float* e) {
    int i = threadIdx.x;
    if (i < NMOL_N) e[i] = 0.0f;
}

__global__ void init_kernel(const float* __restrict__ emb, const int* __restrict__ z,
                            float* __restrict__ s, float* __restrict__ v)
{
    int idx = blockIdx.x * blockDim.x + threadIdx.x;
    if (idx >= N_ATOMS * FDIM) return;
    int n = idx >> 7, g = idx & 127;
    s[idx] = emb[(size_t)z[n] * FDIM + g];
    size_t b = (size_t)n * 384 + g;
    v[b] = 0.0f; v[b + 128] = 0.0f; v[b + 256] = 0.0f;
}

__global__ void vadd_kernel(float* __restrict__ v, const float* __restrict__ vdel) {
    int idx = blockIdx.x * blockDim.x + threadIdx.x;
    if (idx >= N_ATOMS * 3 * FDIM) return;
    v[idx] += vdel[idx];
}

__global__ void vnorm_cat_kernel(const float* __restrict__ s,
                                 const float* __restrict__ vv,
                                 float* __restrict__ cat)
{
    int idx = blockIdx.x * blockDim.x + threadIdx.x;
    if (idx >= N_ATOMS * FDIM) return;
    int n = idx >> 7, g = idx & 127;
    size_t b = (size_t)n * 384 + g;
    float a0 = vv[b], a1 = vv[b + 128], a2 = vv[b + 256];
    cat[(size_t)n * 256 + g]       = s[idx];
    cat[(size_t)n * 256 + 128 + g] = sqrtf(a0*a0 + a1*a1 + a2*a2 + EPS_F);
}

__global__ void update_kernel(const float* __restrict__ uv,
                              const float* __restrict__ vv,
                              const float* __restrict__ aarr,
                              float* __restrict__ s, float* __restrict__ v)
{
    int idx = blockIdx.x * blockDim.x + threadIdx.x;
    if (idx >= N_ATOMS * FDIM) return;
    int n = idx >> 7, g = idx & 127;
    size_t vb = (size_t)n * 384 + g;
    float u0 = uv[vb], u1 = uv[vb + 128], u2 = uv[vb + 256];
    float w0 = vv[vb], w1 = vv[vb + 128], w2 = vv[vb + 256];
    float dot = u0*w0 + u1*w1 + u2*w2;
    float a0 = aarr[vb], a1 = aarr[vb + 128], a2 = aarr[vb + 256];
    s[idx] += a1 * dot + a2;
    v[vb]       += u0 * a0;
    v[vb + 128] += u1 * a0;
    v[vb + 256] += u2 * a0;
}

__global__ void readout_kernel(const float* __restrict__ h3,
                               const float* __restrict__ W2,
                               const float* __restrict__ b2,
                               const int* __restrict__ mol,
                               float* __restrict__ energy)
{
    int n = blockIdx.x * blockDim.x + threadIdx.x;
    if (n >= N_ATOMS) return;
    float acc = b2[0];
    #pragma unroll 8
    for (int g = 0; g < FDIM / 2; ++g)
        acc += h3[(size_t)n * (FDIM / 2) + g] * W2[g];
    atomicAdd(&energy[mol[n]], acc);
}

// ---------------------------------------------------------------------------
static void launch_gemm(const float* A, const unsigned short* Wsw, const float* b,
                        float* C, int M, int K, int Nc, int act, hipStream_t stream)
{
    dim3 grid(M / 16, Nc / 64);
    if (K == 128)
        gemm_wmma_bf16<128><<<grid, 128, 0, stream>>>(A, Wsw, b, C, Nc, act);
    else
        gemm_wmma_bf16<256><<<grid, 128, 0, stream>>>(A, Wsw, b, C, Nc, act);
}

static void launch_swizzle(const float* W, unsigned short* out, int K, int Nc,
                           hipStream_t stream)
{
    int n = K * Nc;
    swizzle_w_kernel<<<(n + 255) / 256, 256, 0, stream>>>(W, out, K, Nc);
}

extern "C" void kernel_launch(void* const* d_in, const int* in_sizes, int n_in,
                              void* d_out, int out_size, void* d_ws, size_t ws_size,
                              hipStream_t stream)
{
    (void)in_sizes; (void)n_in; (void)out_size; (void)ws_size;

    const float* xyz   = (const float*)d_in[0];
    const int*   z     = (const int*)  d_in[1];
    const int*   nbrs  = (const int*)  d_in[2];
    const int*   mol   = (const int*)  d_in[3];
    const float* emb   = (const float*)d_in[4];
    const float* msgW1 = (const float*)d_in[5];
    const float* msgb1 = (const float*)d_in[6];
    const float* msgW2 = (const float*)d_in[7];
    const float* msgb2 = (const float*)d_in[8];
    const float* distW = (const float*)d_in[9];
    const float* distb = (const float*)d_in[10];
    const float* updU  = (const float*)d_in[11];
    const float* updV  = (const float*)d_in[12];
    const float* updW1 = (const float*)d_in[13];
    const float* updb1 = (const float*)d_in[14];
    const float* updW2 = (const float*)d_in[15];
    const float* updb2 = (const float*)d_in[16];
    const float* rW1   = (const float*)d_in[17];
    const float* rb1   = (const float*)d_in[18];
    const float* rW2   = (const float*)d_in[19];
    const float* rb2   = (const float*)d_in[20];
    float* energy = (float*)d_out;

    // ---- workspace partition: fp32 region ----
    float* p    = (float*)d_ws;
    float* s    = p; p += (size_t)N_ATOMS * FDIM;
    float* v    = p; p += (size_t)N_ATOMS * 3 * FDIM;
    float* vdel = p; p += (size_t)N_ATOMS * 3 * FDIM;
    float* h    = p; p += (size_t)N_ATOMS * FDIM;
    float* phi  = p; p += (size_t)N_ATOMS * 3 * FDIM;   // reused for 'a'
    float* uv   = p; p += (size_t)N_ATOMS * 3 * FDIM;
    float* vv   = p; p += (size_t)N_ATOMS * 3 * FDIM;
    float* cat  = p; p += (size_t)N_ATOMS * 2 * FDIM;
    float* rbf  = p; p += (size_t)NEDGE * NRBF_N;
    float* unitv= p; p += (size_t)NEDGE * 3;
    float* fcut = p; p += (size_t)NEDGE;

    // ---- bf16 swizzled-weight region (32B aligned) ----
    uintptr_t up = ((uintptr_t)p + 31) & ~(uintptr_t)31;
    unsigned short* wp = (unsigned short*)up;
    unsigned short *msgW1s[NLAYER], *msgW2s[NLAYER], *updUs[NLAYER], *updVs[NLAYER],
                   *updW1s[NLAYER], *updW2s[NLAYER];
    for (int l = 0; l < NLAYER; ++l) {
        msgW1s[l] = wp; wp += FDIM * FDIM;
        msgW2s[l] = wp; wp += FDIM * 3 * FDIM;
        updUs[l]  = wp; wp += FDIM * FDIM;
        updVs[l]  = wp; wp += FDIM * FDIM;
        updW1s[l] = wp; wp += 2 * FDIM * FDIM;
        updW2s[l] = wp; wp += FDIM * 3 * FDIM;
    }
    unsigned short* rW1s = wp; wp += FDIM * (FDIM / 2);

    // ---- int region ----
    uintptr_t ui = ((uintptr_t)wp + 15) & ~(uintptr_t)15;
    int* ip     = (int*)ui;
    int* deg    = ip; ip += N_ATOMS;
    int* rowptr = ip; ip += N_ATOMS + 1;
    int* cursor = ip; ip += N_ATOMS;
    int* eidx   = ip; ip += NEDGE;

    const int nf  = N_ATOMS * FDIM;
    const int n3f = N_ATOMS * 3 * FDIM;

    // ---- one-time (per launch) prep ----
    zero_energy_kernel<<<1, 128, 0, stream>>>(energy);
    init_kernel<<<(nf + 255) / 256, 256, 0, stream>>>(emb, z, s, v);
    geom_kernel<<<(NEDGE + 255) / 256, 256, 0, stream>>>(xyz, nbrs, rbf, unitv, fcut);
    zero_int_kernel<<<(N_ATOMS + 255) / 256, 256, 0, stream>>>(deg, N_ATOMS);
    hist_kernel<<<(NEDGE + 255) / 256, 256, 0, stream>>>(nbrs, deg);
    scan_kernel<<<1, 256, 0, stream>>>(deg, rowptr, cursor);
    fill_kernel<<<(NEDGE + 255) / 256, 256, 0, stream>>>(nbrs, cursor, eidx);
    for (int l = 0; l < NLAYER; ++l) {
        launch_swizzle(msgW1 + (size_t)l*FDIM*FDIM,     msgW1s[l], FDIM,   FDIM,   stream);
        launch_swizzle(msgW2 + (size_t)l*FDIM*3*FDIM,   msgW2s[l], FDIM,   3*FDIM, stream);
        launch_swizzle(updU  + (size_t)l*FDIM*FDIM,     updUs[l],  FDIM,   FDIM,   stream);
        launch_swizzle(updV  + (size_t)l*FDIM*FDIM,     updVs[l],  FDIM,   FDIM,   stream);
        launch_swizzle(updW1 + (size_t)l*2*FDIM*FDIM,   updW1s[l], 2*FDIM, FDIM,   stream);
        launch_swizzle(updW2 + (size_t)l*FDIM*3*FDIM,   updW2s[l], FDIM,   3*FDIM, stream);
    }
    launch_swizzle(rW1, rW1s, FDIM, FDIM / 2, stream);

    // ---- layers ----
    for (int l = 0; l < NLAYER; ++l) {
        launch_gemm(s, msgW1s[l], msgb1 + (size_t)l * FDIM,
                    h, N_ATOMS, FDIM, FDIM, 1, stream);
        launch_gemm(h, msgW2s[l], msgb2 + (size_t)l * 3 * FDIM,
                    phi, N_ATOMS, FDIM, 3 * FDIM, 0, stream);
        node_gather_kernel<<<N_ATOMS, 128, 0, stream>>>(
            rowptr, eidx, nbrs, rbf, unitv, fcut,
            distW + (size_t)l * NRBF_N * 3 * FDIM, distb + (size_t)l * 3 * FDIM,
            phi, v, s, vdel);
        vadd_kernel<<<(n3f + 255) / 256, 256, 0, stream>>>(v, vdel);
        launch_gemm(v, updUs[l], nullptr, uv, 3 * N_ATOMS, FDIM, FDIM, 0, stream);
        launch_gemm(v, updVs[l], nullptr, vv, 3 * N_ATOMS, FDIM, FDIM, 0, stream);
        vnorm_cat_kernel<<<(nf + 255) / 256, 256, 0, stream>>>(s, vv, cat);
        launch_gemm(cat, updW1s[l], updb1 + (size_t)l * FDIM,
                    h, N_ATOMS, 2 * FDIM, FDIM, 1, stream);
        launch_gemm(h, updW2s[l], updb2 + (size_t)l * 3 * FDIM,
                    phi, N_ATOMS, FDIM, 3 * FDIM, 0, stream);
        update_kernel<<<(nf + 255) / 256, 256, 0, stream>>>(uv, vv, phi, s, v);
    }

    // ---- readout ----
    launch_gemm(s, rW1s, rb1, h, N_ATOMS, FDIM, FDIM / 2, 1, stream);
    readout_kernel<<<(N_ATOMS + 255) / 256, 256, 0, stream>>>(h, rW2, rb2, mol, energy);
}
